// DomainContrastiveLoss_17068200034933
// MI455X (gfx1250) — matrix operand order repaired and checked by reading, not compiled, not run on hardware
//
#include <hip/hip_runtime.h>

// ---------------------------------------------------------------------------
// CDNA5 (gfx1250, wave32) implementation of DomainContrastiveLoss.
// GEMMs run on v_wmma_f32_16x16x32_bf16 with an f32->bf16 hi/lo split
// (3 WMMAs per tile: hi*hi + hi*lo + lo*hi) for near-f32 accuracy at the
// fast bf16 matrix rate. Working set (<80MB) is L2-resident (192MB L2),
// so fragments are loaded straight from global per the ISA 7.12.2 layouts.
//
// v2: templated D/Npad -> all tile offsets become 24-bit instruction
// immediates; 6 hoisted flat pointers advance +64B/K-step (kills the
// per-tile v_add_nc_u64/v_lshlrev_b64 chains and the WMMA->VALU v_nop
// hazards); WMMAs grouped 3x8 independent ops so each accumulator's
// reuse distance is 8 (hides XDL latency).
// ---------------------------------------------------------------------------

typedef __bf16 bf16;
typedef __bf16 bf16x16 __attribute__((ext_vector_type(16)));
typedef float  f32x8   __attribute__((ext_vector_type(8)));
typedef unsigned int u32x4 __attribute__((ext_vector_type(4)));

union Frag { bf16x16 v; u32x4 u[2]; };

#define WMMA_BF16(A, B, Cacc) \
  __builtin_amdgcn_wmma_f32_16x16x32_bf16(false, (A), false, (B), (short)0, (Cacc), false, false)

// ----------------------------- block reductions ----------------------------

__device__ __forceinline__ float block_reduce_sum(float v, volatile float* sh) {
  #pragma unroll
  for (int o = 16; o > 0; o >>= 1) v += __shfl_xor(v, o, 32);
  const int wave = threadIdx.x >> 5;
  const int lane = threadIdx.x & 31;
  __syncthreads();                       // protect shared buffer reuse
  if (lane == 0) sh[wave] = v;
  __syncthreads();
  if (threadIdx.x == 0) {
    float t = 0.0f;
    const int nw = (int)(blockDim.x >> 5);
    for (int i = 0; i < nw; ++i) t += sh[i];
    sh[0] = t;
  }
  __syncthreads();
  return sh[0];
}

__device__ __forceinline__ float block_reduce_max(float v, volatile float* sh) {
  #pragma unroll
  for (int o = 16; o > 0; o >>= 1) v = fmaxf(v, __shfl_xor(v, o, 32));
  const int wave = threadIdx.x >> 5;
  const int lane = threadIdx.x & 31;
  __syncthreads();
  if (lane == 0) sh[wave] = v;
  __syncthreads();
  if (threadIdx.x == 0) {
    float t = -3.4e38f;
    const int nw = (int)(blockDim.x >> 5);
    for (int i = 0; i < nw; ++i) t = fmaxf(t, sh[i]);
    sh[0] = t;
  }
  __syncthreads();
  return sh[0];
}

// ----------------------- row L2 normalize + bf16 split ---------------------

__global__ void __launch_bounds__(256)
normalize_split_kernel(const float* __restrict__ in,
                       bf16* __restrict__ hi, bf16* __restrict__ lo, int D) {
  __shared__ float sh[8];
  const int row = blockIdx.x;
  const float* x = in + (size_t)row * D;
  float s = 0.0f;
  for (int j = threadIdx.x; j < D; j += blockDim.x) { float v = x[j]; s += v * v; }
  const float tot   = block_reduce_sum(s, sh);
  const float scale = 1.0f / fmaxf(sqrtf(tot), 1e-12f);
  bf16* ho = hi + (size_t)row * D;
  bf16* lw = lo + (size_t)row * D;
  for (int j = threadIdx.x; j < D; j += blockDim.x) {
    const float y = x[j] * scale;
    const bf16 h = (bf16)y;
    ho[j] = h;
    lw[j] = (bf16)(y - (float)h);
  }
}

// ------------------------- split-bf16 WMMA GEMM ----------------------------
// S[M, Npad] = A[M, D] * B[Npad, D]^T  (A, B pre-split into hi/lo bf16).
// Block = 256 threads = 8 waves, arranged 4(M) x 2(N); WG tile 128x128.
// Each wave computes a 32x64 tile = 2x4 WMMA accumulators.
// DD/NN > 0: compile-time D / Npad (all tile offsets fold to immediates).

template<int DD, int NN>
__global__ void __launch_bounds__(256)
gemm_bf16x2_kernel(const bf16* __restrict__ Ah, const bf16* __restrict__ Al,
                   const bf16* __restrict__ Bh, const bf16* __restrict__ Bl,
                   float* __restrict__ S, int npad_rt, int d_rt) {
  const int D    = (DD > 0) ? DD : d_rt;
  const int Npad = (NN > 0) ? NN : npad_rt;

  const int tid  = threadIdx.x;
  const int wave = tid >> 5;
  const int lane = tid & 31;
  const int lm   = lane & 15;   // row within 16-row fragment
  const int lg   = lane >> 4;   // lane group (K interleave select)

  const int wm    = wave & 3;   // 0..3 along M
  const int wn    = wave >> 2;  // 0..1 along N
  const int mbase = blockIdx.y * 128 + wm * 32;
  const int nbase = blockIdx.x * 128 + wn * 64;

  f32x8 acc[2][4] = {};

  // Hoisted flat pointers; each advances +32 elements (+64B) per K-step.
  // A fragment (16x32 bf16, ISA 7.12.2): lane group 0 -> K {0..7,16..23},
  // group 1 -> K {8..15,24..31}: chunks at +lg*8 and +lg*8+16 elements.
  const bf16* pa0h = Ah + (size_t)(mbase +      lm) * D + lg * 8;
  const bf16* pa1h = Ah + (size_t)(mbase + 16 + lm) * D + lg * 8;
  const bf16* pa0l = Al + (size_t)(mbase +      lm) * D + lg * 8;
  const bf16* pa1l = Al + (size_t)(mbase + 16 + lm) * D + lg * 8;
  // B fragment (32x16, K x N): lane group 0 -> K {0..7,16..23}, group 1 -> K {8..15,24..31},
  // N = lane&15 -> proto row. Chunks at +lg*8 and +lg*8+16 elements.
  const size_t brow = (size_t)(nbase + lm) * D + lg * 8;
  const bf16* pbh = Bh + brow;
  const bf16* pbl = Bl + brow;

  for (int k = 0; k < D; k += 32) {
    Frag ah[2], al[2], bh[4], bl[4];
    ah[0].u[0] = *(const u32x4*)(pa0h); ah[0].u[1] = *(const u32x4*)(pa0h + 16);
    ah[1].u[0] = *(const u32x4*)(pa1h); ah[1].u[1] = *(const u32x4*)(pa1h + 16);
    al[0].u[0] = *(const u32x4*)(pa0l); al[0].u[1] = *(const u32x4*)(pa0l + 16);
    al[1].u[0] = *(const u32x4*)(pa1l); al[1].u[1] = *(const u32x4*)(pa1l + 16);
    #pragma unroll
    for (int nt = 0; nt < 4; ++nt) {
      const bf16* p = pbh + (size_t)nt * 16 * D;   // const offset when DD>0
      bh[nt].u[0] = *(const u32x4*)(p);
      bh[nt].u[1] = *(const u32x4*)(p + 16);
    }
    #pragma unroll
    for (int nt = 0; nt < 4; ++nt) {
      const bf16* p = pbl + (size_t)nt * 16 * D;
      bl[nt].u[0] = *(const u32x4*)(p);
      bl[nt].u[1] = *(const u32x4*)(p + 16);
    }

    // Three groups of 8 independent WMMAs: same-acc reuse distance = 8.
    #pragma unroll
    for (int nt = 0; nt < 4; ++nt)
      #pragma unroll
      for (int mt = 0; mt < 2; ++mt)
        acc[mt][nt] = WMMA_BF16(ah[mt].v, bh[nt].v, acc[mt][nt]);   // hi*hi
    #pragma unroll
    for (int nt = 0; nt < 4; ++nt)
      #pragma unroll
      for (int mt = 0; mt < 2; ++mt)
        acc[mt][nt] = WMMA_BF16(al[mt].v, bh[nt].v, acc[mt][nt]);   // lo*hi
    #pragma unroll
    for (int nt = 0; nt < 4; ++nt)
      #pragma unroll
      for (int mt = 0; mt < 2; ++mt)
        acc[mt][nt] = WMMA_BF16(ah[mt].v, bl[nt].v, acc[mt][nt]);   // hi*lo

    pa0h += 32; pa1h += 32; pa0l += 32; pa1l += 32; pbh += 32; pbl += 32;
  }

  // C/D layout: VGPR r, lanes 0-15 -> M=r, lanes 16-31 -> M=8+r; N=lane&15.
  float* sp = S + (size_t)(mbase + lg * 8) * Npad + nbase + lm;
  #pragma unroll
  for (int mt = 0; mt < 2; ++mt)
    #pragma unroll
    for (int nt = 0; nt < 4; ++nt)
      #pragma unroll
      for (int r = 0; r < 8; ++r)
        sp[(size_t)(mt * 16 + r) * Npad + nt * 16] = acc[mt][nt][r];
}

// -------------- intra-domain InfoNCE + positive-alignment terms ------------

__global__ void __launch_bounds__(256)
row_infonce_kernel(const float* __restrict__ S, const int* __restrict__ labels,
                   int C, int Npad,
                   float* __restrict__ pos_out, float* __restrict__ inf_out) {
  __shared__ float sh[8];
  const int row = blockIdx.x;
  const float* x = S + (size_t)row * Npad;
  float s = 0.0f, s2 = 0.0f, mx = -3.4e38f;
  for (int j = threadIdx.x; j < C; j += blockDim.x) {
    const float v = x[j];
    s += v; s2 += v * v; mx = fmaxf(mx, v);
  }
  const float sum   = block_reduce_sum(s, sh);
  const float sumsq = block_reduce_sum(s2, sh);
  const float vmax  = block_reduce_max(mx, sh);
  const float mean  = sum / (float)C;
  const float var   = fmaxf((sumsq - (float)C * mean * mean) / (float)(C - 1), 0.0f);
  const float temp  = fminf(fmaxf(0.07f * (1.0f + sqrtf(var)), 0.01f), 0.5f);
  float e = 0.0f;
  for (int j = threadIdx.x; j < C; j += blockDim.x)
    e += expf((x[j] - vmax) / temp);
  const float sume = block_reduce_sum(e, sh);
  if (threadIdx.x == 0) {
    const float pos = x[labels[row]];           // pos_cos == sims[i, label]
    const float lse = vmax / temp + logf(sume);
    pos_out[row] = 1.0f - pos;
    inf_out[row] = lse - pos / temp;
  }
}

// ---------------- cross-domain masked stats + logsumexp --------------------

__global__ void __launch_bounds__(256)
row_cross_kernel(const float* __restrict__ S, const int* __restrict__ labels,
                 const int* __restrict__ plab, int P, int Npad,
                 float* __restrict__ cross_out) {
  __shared__ float sh[8];
  __shared__ int spl[2048];
  const int row = blockIdx.x;
  const int lab = labels[row];
  for (int j = threadIdx.x; j < P; j += blockDim.x) spl[j] = plab[j];
  __syncthreads();
  const float* x = S + (size_t)row * Npad;
  float s = 0.0f, s2 = 0.0f, mx = -3.4e38f, cnt = 0.0f;
  for (int j = threadIdx.x; j < P; j += blockDim.x) {
    if (spl[j] != lab) {
      const float v = x[j];
      s += v; s2 += v * v; mx = fmaxf(mx, v); cnt += 1.0f;
    }
  }
  const float n     = block_reduce_sum(cnt, sh);
  const float sum   = block_reduce_sum(s, sh);
  const float sumsq = block_reduce_sum(s2, sh);
  const float vmax  = block_reduce_max(mx, sh);
  const float mean  = sum / n;
  const float var   = fmaxf((sumsq - n * mean * mean) / (n - 1.0f), 0.0f);
  const float t     = fminf(fmaxf(0.07f * (1.0f + sqrtf(var)), 0.01f), 0.5f) * 2.0f;
  float e = 0.0f;
  for (int j = threadIdx.x; j < P; j += blockDim.x)
    if (spl[j] != lab) e += expf((x[j] - vmax) / t);
  const float sume = block_reduce_sum(e, sh);
  if (threadIdx.x == 0) cross_out[row] = vmax / t + logf(sume);
}

// -------------------- deterministic final reduction ------------------------

__global__ void __launch_bounds__(256)
finalize_kernel(const float* __restrict__ pos_a, const float* __restrict__ inf_a,
                const float* __restrict__ cross_a, int B, float* __restrict__ out) {
  __shared__ float sh[8];
  float p = 0.0f, q = 0.0f, c = 0.0f;
  for (int j = threadIdx.x; j < B; j += blockDim.x) {
    p += pos_a[j]; q += inf_a[j]; c += cross_a[j];
  }
  const float ps = block_reduce_sum(p, sh);
  const float qs = block_reduce_sum(q, sh);
  const float cs = block_reduce_sum(c, sh);
  if (threadIdx.x == 0) {
    const float invB      = 1.0f / (float)B;
    const float w         = 0.1f + 0.9f * (1.0f / 1000.0f);   // STEP/WARMUP
    const float curr_beta = 0.5f * w;                          // BETA * w
    out[0] = ps * invB + curr_beta * (qs * invB + 0.3f * (cs * invB));
  }
}

// ------------------------------- launcher ----------------------------------

extern "C" void kernel_launch(void* const* d_in, const int* in_sizes, int n_in,
                              void* d_out, int out_size, void* d_ws, size_t ws_size,
                              hipStream_t stream) {
  (void)n_in; (void)out_size;
  const float* features   = (const float*)d_in[0];
  const float* cur_proto  = (const float*)d_in[1];
  const float* prev_proto = (const float*)d_in[2];
  const int*   labels     = (const int*)d_in[3];
  const int*   plabels    = (const int*)d_in[4];

  const int B = in_sizes[3];            // 4096
  const int P = in_sizes[4];            // 2000
  const int D = in_sizes[0] / B;        // 1024
  const int C = in_sizes[1] / D;        // 1000
  const int Cpad = ((C + 127) / 128) * 128;   // 1024
  const int Ppad = ((P + 127) / 128) * 128;   // 2048

  char* ws = (char*)d_ws;
  size_t off = 0;
  auto take = [&](size_t bytes) -> char* {
    char* p = ws + off;
    off += (bytes + 255) & ~(size_t)255;
    return p;
  };

  bf16* fh  = (bf16*)take((size_t)B * D * sizeof(bf16));
  bf16* fl  = (bf16*)take((size_t)B * D * sizeof(bf16));
  bf16* cph = (bf16*)take((size_t)Cpad * D * sizeof(bf16));
  bf16* cpl = (bf16*)take((size_t)Cpad * D * sizeof(bf16));
  bf16* pph = (bf16*)take((size_t)Ppad * D * sizeof(bf16));
  bf16* ppl = (bf16*)take((size_t)Ppad * D * sizeof(bf16));
  float* sims   = (float*)take((size_t)B * Cpad * sizeof(float));
  float* csims  = (float*)take((size_t)B * Ppad * sizeof(float));
  float* pos_a  = (float*)take((size_t)B * sizeof(float));
  float* inf_a  = (float*)take((size_t)B * sizeof(float));
  float* cross_a= (float*)take((size_t)B * sizeof(float));
  (void)ws_size;  // ~80 MB needed; assume workspace is large enough

  // Zero proto buffers so padded rows (C..Cpad-1, P..Ppad-1) contribute 0.
  hipMemsetAsync(cph, 0, (size_t)Cpad * D * sizeof(bf16), stream);
  hipMemsetAsync(cpl, 0, (size_t)Cpad * D * sizeof(bf16), stream);
  hipMemsetAsync(pph, 0, (size_t)Ppad * D * sizeof(bf16), stream);
  hipMemsetAsync(ppl, 0, (size_t)Ppad * D * sizeof(bf16), stream);

  normalize_split_kernel<<<B, 256, 0, stream>>>(features,   fh,  fl,  D);
  normalize_split_kernel<<<C, 256, 0, stream>>>(cur_proto,  cph, cpl, D);
  normalize_split_kernel<<<P, 256, 0, stream>>>(prev_proto, pph, ppl, D);

  const dim3 g1(Cpad / 128, B / 128), g2(Ppad / 128, B / 128);
  if (D == 1024 && Cpad == 1024)
    gemm_bf16x2_kernel<1024, 1024><<<g1, 256, 0, stream>>>(fh, fl, cph, cpl, sims, Cpad, D);
  else
    gemm_bf16x2_kernel<-1, -1><<<g1, 256, 0, stream>>>(fh, fl, cph, cpl, sims, Cpad, D);
  if (D == 1024 && Ppad == 2048)
    gemm_bf16x2_kernel<1024, 2048><<<g2, 256, 0, stream>>>(fh, fl, pph, ppl, csims, Ppad, D);
  else
    gemm_bf16x2_kernel<-1, -1><<<g2, 256, 0, stream>>>(fh, fl, pph, ppl, csims, Ppad, D);

  row_infonce_kernel<<<B, 256, 0, stream>>>(sims, labels, C, Cpad, pos_a, inf_a);
  row_cross_kernel<<<B, 256, 0, stream>>>(csims, labels, plabels, P, Ppad, cross_a);

  finalize_kernel<<<1, 256, 0, stream>>>(pos_a, inf_a, cross_a, B, (float*)d_out);
}
